// MoE4Embedder_7988639170560
// MI455X (gfx1250) — matrix operand order repaired and checked by reading, not compiled, not run on hardware
//
#include <hip/hip_runtime.h>
#include <hip/hip_bf16.h>

// ---------------------------------------------------------------------------
// MoE gene embedder fused kernel for gfx1250 (MI455X).
// Dominant GEMM (16000x512 @ 512x512) runs on v_wmma_f32_16x16x32_bf16.
// Memory-bound target: ~65MB HBM traffic -> ~2.8us at 23.3 TB/s.
// ---------------------------------------------------------------------------

#define Bdim 32
#define Tdim 500
#define Ddim 512
#define NUM_SHARED 5
#define NUM_ROUTING 10
#define TOPK 5
#define ROWS (Bdim * Tdim)          // 16000
#define MTILE 16
#define NBLOCKS (ROWS / MTILE)      // 1000

typedef __attribute__((ext_vector_type(8)))  float        v8f;
typedef __attribute__((ext_vector_type(8)))  unsigned int u32x8;
typedef __attribute__((ext_vector_type(16))) __bf16       v16bf;

__device__ __forceinline__ unsigned short f32_to_bf16_rne(float f) {
    unsigned int u = __float_as_uint(f);
    unsigned int r = u + 0x7FFFu + ((u >> 16) & 1u);   // round-to-nearest-even
    return (unsigned short)(r >> 16);
}

// --- Kernel 1: convert router_w1 (f32 [512,512]) to bf16 in workspace -------
__global__ void convert_w1_bf16(const float* __restrict__ w1,
                                unsigned short* __restrict__ w1bf) {
    int idx = blockIdx.x * blockDim.x + threadIdx.x;
    const int total = Ddim * Ddim;
    for (int i = idx; i < total; i += gridDim.x * blockDim.x)
        w1bf[i] = f32_to_bf16_rne(w1[i]);
}

// --- Kernel 2: fused MoE per 16-row tile ------------------------------------
__global__ __launch_bounds__(256)
void moe_fused_kernel(const float* __restrict__ gene,        // [ROWS, D]
                      const float* __restrict__ value,       // [ROWS]
                      const float* __restrict__ shared_w,    // [5, D]
                      const float* __restrict__ routing_w,   // [10, D]
                      const unsigned short* __restrict__ w1bf, // [D, D] bf16
                      const float* __restrict__ w2,          // [10, D]
                      float* __restrict__ out) {             // [ROWS, D]
    // LDS. u_mem is time-multiplexed: X tile (bf16, 16KB) during the GEMM,
    // routing_w (f32, 20KB) during the output phase.
    __shared__ __align__(16) unsigned char u_mem[NUM_ROUTING * Ddim * 4]; // 20KB
    __shared__ float h_lds[MTILE * Ddim];                                 // 32KB
    __shared__ float shared_sum[Ddim];                                    // 2KB
    __shared__ float logits_lds[MTILE * NUM_ROUTING];
    __shared__ float coeff_lds[MTILE * NUM_ROUTING];
    __shared__ float vals_lds[MTILE];

    unsigned short* x_lds  = (unsigned short*)u_mem;   // [16][512] bf16
    float*          rw_lds = (float*)u_mem;            // [10][512] f32

    const int tid  = threadIdx.x;
    const int lane = tid & 31;
    const int wv   = tid >> 5;          // 8 waves; wave wv owns N in [64wv,64wv+64)
    const int half = lane >> 4;         // hi/lo half of the wave
    const int l16  = lane & 15;
    const int row0 = blockIdx.x * MTILE;

    // ---- Phase A: stage X tile into LDS as bf16 (coalesced) ----
    for (int i = tid; i < MTILE * Ddim; i += 256) {
        int r = i >> 9, c = i & (Ddim - 1);
        x_lds[i] = f32_to_bf16_rne(gene[(size_t)(row0 + r) * Ddim + c]);
    }
    __syncthreads();

    // ---- Phase B: h = relu(X @ W1^T) via bf16 WMMA, f32 accumulate ----
    v8f acc[4];
    #pragma unroll
    for (int nt = 0; nt < 4; ++nt) acc[nt] = (v8f){0.f,0.f,0.f,0.f,0.f,0.f,0.f,0.f};

    const unsigned int* w1p = (const unsigned int*)w1bf;   // packed bf16 pairs

    for (int k0 = 0; k0 < Ddim; k0 += 32) {
        // A fragment: 16x32 bf16 per ISA layout (lane half selects K groups).
        u32x8 ai;
        #pragma unroll
        for (int j = 0; j < 8; ++j) {
            int kk = ((j < 4) ? (2 * j) : (16 + 2 * (j - 4))) + 8 * half;
            ai[j] = *(const unsigned int*)&x_lds[l16 * Ddim + k0 + kk];
        }
        v16bf a = __builtin_bit_cast(v16bf, ai);

        #pragma unroll
        for (int nt = 0; nt < 4; ++nt) {
            // B fragment: 32x16 bf16; lane = N column, K split by lane half.
            int n  = wv * 64 + nt * 16 + l16;
            int kb = k0 + half * 8;
            u32x8 bi;
            #pragma unroll
            for (int j = 0; j < 8; ++j)
                bi[j] = w1p[(n * Ddim + kb + ((j < 4) ? (2 * j) : (16 + 2 * (j - 4)))) >> 1];
            v16bf b = __builtin_bit_cast(v16bf, bi);

            acc[nt] = __builtin_amdgcn_wmma_f32_16x16x32_bf16(
                false, a, false, b, (short)0, acc[nt], false, false);
        }
    }

    // ReLU + scatter C fragments to h LDS (C layout: VGPR i -> M=i+8*half).
    #pragma unroll
    for (int nt = 0; nt < 4; ++nt) {
        int col = wv * 64 + nt * 16 + l16;
        #pragma unroll
        for (int i = 0; i < 8; ++i) {
            int r = i + 8 * half;
            float v = acc[nt][i];
            h_lds[r * Ddim + col] = v > 0.f ? v : 0.f;
        }
    }
    __syncthreads();

    // ---- Phase C: logits + stage routing_w / shared_sum / values ----
    if (tid < MTILE * NUM_ROUTING) {            // 160 threads: one (row,e) each
        int r = tid / NUM_ROUTING, e = tid % NUM_ROUTING;
        const float* w2r = w2 + (size_t)e * Ddim;
        float s = 0.f;
        for (int k = 0; k < Ddim; ++k) s += h_lds[r * Ddim + k] * w2r[k];
        logits_lds[tid] = s;
    }
    for (int i = tid; i < NUM_ROUTING * Ddim; i += 256) rw_lds[i] = routing_w[i];
    for (int d = tid; d < Ddim; d += 256) {
        float s = 0.f;
        #pragma unroll
        for (int e = 0; e < NUM_SHARED; ++e) s += shared_w[e * Ddim + d];
        shared_sum[d] = s;
    }
    if (tid < MTILE) vals_lds[tid] = value[row0 + tid];
    __syncthreads();

    // ---- Phase D: softmax + top-5 mask (16 threads, one row each) ----
    if (tid < MTILE) {
        float lg[NUM_ROUTING];
        float mx = -3.4e38f;
        #pragma unroll
        for (int e = 0; e < NUM_ROUTING; ++e) {
            lg[e] = logits_lds[tid * NUM_ROUTING + e];
            mx = fmaxf(mx, lg[e]);
        }
        float ex[NUM_ROUTING], s = 0.f;
        #pragma unroll
        for (int e = 0; e < NUM_ROUTING; ++e) { ex[e] = __expf(lg[e] - mx); s += ex[e]; }
        float inv = 1.f / s;
        #pragma unroll
        for (int e = 0; e < NUM_ROUTING; ++e) {
            int rank = 0;   // deterministic tie-break: earlier index wins (top_k)
            #pragma unroll
            for (int e2 = 0; e2 < NUM_ROUTING; ++e2)
                rank += (lg[e2] > lg[e]) || (lg[e2] == lg[e] && e2 < e);
            coeff_lds[tid * NUM_ROUTING + e] = (rank < TOPK) ? ex[e] * inv : 0.f;
        }
    }
    __syncthreads();

    // ---- Phase E: out = v * (shared_sum + sparse_w @ routing_w) ----
    for (int r = 0; r < MTILE; ++r) {
        float v = vals_lds[r];
        const float* cf = &coeff_lds[r * NUM_ROUTING];
        #pragma unroll
        for (int pass = 0; pass < 2; ++pass) {
            int d = tid + pass * 256;
            float a0 = shared_sum[d];
            #pragma unroll
            for (int e = 0; e < NUM_ROUTING; ++e)
                a0 += cf[e] * rw_lds[e * Ddim + d];
            out[(size_t)(row0 + r) * Ddim + d] = v * a0;
        }
    }
}

extern "C" void kernel_launch(void* const* d_in, const int* in_sizes, int n_in,
                              void* d_out, int out_size, void* d_ws, size_t ws_size,
                              hipStream_t stream) {
    const float* gene      = (const float*)d_in[0];   // [32,500,512]
    const float* value     = (const float*)d_in[1];   // [32,500]
    const float* shared_w  = (const float*)d_in[2];   // [5,512]
    const float* routing_w = (const float*)d_in[3];   // [10,512]
    const float* router_w1 = (const float*)d_in[4];   // [512,512]
    const float* router_w2 = (const float*)d_in[5];   // [10,512]
    float* out = (float*)d_out;

    unsigned short* w1bf = (unsigned short*)d_ws;     // 512KB scratch

    convert_w1_bf16<<<256, 256, 0, stream>>>(router_w1, w1bf);
    moe_fused_kernel<<<NBLOCKS, 256, 0, stream>>>(
        gene, value, shared_w, routing_w, w1bf, router_w2, out);
}
